// GSA_47373489275249
// MI455X (gfx1250) — matrix operand order repaired and checked by reading, not compiled, not run on hardware
//
#include <hip/hip_runtime.h>
#include <hip/hip_bf16.h>

typedef _Float16 h16;
typedef __attribute__((ext_vector_type(16))) _Float16 v16h;
typedef __attribute__((ext_vector_type(8)))  _Float16 v8h;
typedef __attribute__((ext_vector_type(8)))  float    v8f;
typedef __attribute__((ext_vector_type(4)))  unsigned int u32x4;
typedef __attribute__((ext_vector_type(8)))  int      i32x8;
typedef __attribute__((ext_vector_type(4)))  int      i32x4;

#define NPIX 200704   // 448*448
#define CCH  192
#define NHD  6
#define HD   32

#define HAVE_TDM __has_builtin(__builtin_amdgcn_tensor_load_to_lds)

// ---------------- small utility kernels ----------------
__global__ void cvt_f32_to_f16(const float* __restrict__ s, h16* __restrict__ d, int n) {
    int i = blockIdx.x * 256 + threadIdx.x;
    if (i < n) d[i] = (h16)s[i];
}

__global__ void zero_f32(float* __restrict__ p, int n) {
    int i = blockIdx.x * 256 + threadIdx.x;
    if (i < n) p[i] = 0.f;
}

// ---------------- channel LayerNorm (per pixel over C=192) ----------------
__global__ void layernorm_kernel(const float* __restrict__ x, const float* __restrict__ g,
                                 const float* __restrict__ b, h16* __restrict__ xn) {
    __shared__ float sg[CCH], sb[CCH];
    for (int i = threadIdx.x; i < CCH; i += blockDim.x) { sg[i] = g[i]; sb[i] = b[i]; }
    __syncthreads();
    int p = blockIdx.x * blockDim.x + threadIdx.x;   // NPIX = 784*256, no tail
    float s = 0.f, ss = 0.f;
    for (int c = 0; c < CCH; ++c) {
        float v = x[(size_t)c * NPIX + p];
        s += v; ss += v * v;
    }
    float mu  = s * (1.0f / CCH);
    float var = ss * (1.0f / CCH) - mu * mu;
    float rs  = rsqrtf(var + 1e-5f);
    for (int c = 0; c < CCH; ++c) {
        float v = (x[(size_t)c * NPIX + p] - mu) * rs * sg[c] + sb[c];
        xn[(size_t)c * NPIX + p] = (h16)v;
    }
}

#if HAVE_TDM
// Build a 2D TDM descriptor: 32(K-rows) x 64(N-cols) f16 tile, row stride NPIX elems,
// from global 'gsrc' into LDS byte offset 'lds_off'. Fields per CDNA5 ISA 8.3/8.4.
__device__ __forceinline__ void tdm_load_b_tile(const h16* gsrc, unsigned lds_off, int krows) {
    unsigned long long ga = (unsigned long long)(uintptr_t)gsrc;
    u32x4 g0;
    g0.x = 1u;                                            // count=1 (valid descriptor)
    g0.y = lds_off;                                       // lds_addr (bytes)
    g0.z = (unsigned)ga;                                  // global_addr[31:0]
    g0.w = (unsigned)((ga >> 32) & 0x01ffffffull)         // global_addr[56:32]
         | (2u << 30);                                    // type=2 ("image")
    i32x8 g1;
    g1[0] = 0x00010000;                                   // wg_mask=0, data_size=1 (2 bytes)
    g1[1] = (int)(((unsigned)NPIX & 0xffffu) << 16);      // tensor_dim0[15:0]
    g1[2] = (int)(((unsigned)NPIX >> 16)                  // tensor_dim0[31:16]
         | (((unsigned)krows & 0xffffu) << 16));          // tensor_dim1[15:0]
    g1[3] = (int)(64u << 16);                             // tensor_dim1[31:16]=0 | tile_dim0=64
    g1[4] = 32;                                           // tile_dim1=32 | tile_dim2=0
    g1[5] = NPIX;                                         // tensor_dim0_stride[31:0]
    g1[6] = 0;                                            // stride0[47:32] | stride1[15:0]
    g1[7] = 0;                                            // stride1[47:16]
    i32x4 z4 = {};
#if __clang_major__ >= 23
    i32x8 z8 = {};
    __builtin_amdgcn_tensor_load_to_lds(g0, g1, z4, z4, z8, 0);
#else
    __builtin_amdgcn_tensor_load_to_lds(g0, g1, z4, z4, 0);
#endif
}
#endif

// ---------------- WMMA GEMM: C[M,N] = A[M,K] * B[K,N] ----------------
// MODE 0: f16 out. MODE 1: +bias[row], f16 out. MODE 2: +resid[row,col], f32 out.
// Block: 256 thr = 8 waves; tile 64(M) x 64(N); wave (wm 0..3, wn 0..1) owns 16x32.
template<int MODE>
__global__ void gemm_wmma(const h16* __restrict__ A, const h16* __restrict__ B,
                          const float* __restrict__ aux, void* __restrict__ outp,
                          int K) {
    __shared__ h16 Bs[32 * 64];
    const int tid  = threadIdx.x;
    const int lane = tid & 31, l = lane & 15, hi = lane >> 4;
    const int w    = tid >> 5,  wm = w & 3,  wn = w >> 2;
    const int m0   = blockIdx.y * 64 + wm * 16;
    const int nblk = blockIdx.x * 64;
    const int brow = tid >> 3, bcol = (tid & 7) * 8;
    (void)brow; (void)bcol;

    v8f c0 = {}, c1 = {};
    for (int k0 = 0; k0 < K; k0 += 32) {
#if HAVE_TDM
        // Tensor Data Mover stages the 32x64 B tile; wave 0 issues + waits TENSORcnt.
        if (w == 0) {
            tdm_load_b_tile(&B[(size_t)k0 * NPIX + nblk], (unsigned)(uintptr_t)Bs, K);
            __builtin_amdgcn_s_wait_tensorcnt(0);
        }
        __syncthreads();
#else
        // stage 32x64 B tile into LDS (coalesced b128 loads)
        *(v8h*)&Bs[brow * 64 + bcol] =
            *(const v8h*)&B[(size_t)(k0 + brow) * NPIX + nblk + bcol];
        if (k0 + 32 < K)
            __builtin_prefetch(&B[(size_t)(k0 + 32 + brow) * NPIX + nblk + bcol], 0, 1);
        __syncthreads();
#endif

        // A fragment: lane<16 -> K k0+0..15 ; lane>=16 -> k0+16..31
        v16h a;
        {
            const v8h alo = *(const v8h*)&A[(size_t)(m0 + l) * K + k0 + hi * 16];
            const v8h ahi = *(const v8h*)&A[(size_t)(m0 + l) * K + k0 + hi * 16 + 8];
            #pragma unroll
            for (int i = 0; i < 8; ++i) { a[i] = alo[i]; a[i + 8] = ahi[i]; }
        }
        // B fragments: lane holds column (lane&15); lanes0-15 K 0..15, lanes16-31 K 16..31
        v16h b0, b1;
        #pragma unroll
        for (int j = 0; j < 16; ++j) {
            b0[j] = Bs[(hi * 16 + j) * 64 + wn * 32 + l];
            b1[j] = Bs[(hi * 16 + j) * 64 + wn * 32 + 16 + l];
        }
        c0 = __builtin_amdgcn_wmma_f32_16x16x32_f16(false, a, false, b0, (short)0, c0, false, false);
        c1 = __builtin_amdgcn_wmma_f32_16x16x32_f16(false, a, false, b1, (short)0, c1, false, false);
        __syncthreads();
    }

    #pragma unroll
    for (int r = 0; r < 8; ++r) {
        const int row = m0 + r + hi * 8;
        const int col = nblk + wn * 32 + l;
        if (MODE == 2) {
            float* o = (float*)outp;
            o[(size_t)row * NPIX + col]      = c0[r] + aux[(size_t)row * NPIX + col];
            o[(size_t)row * NPIX + col + 16] = c1[r] + aux[(size_t)row * NPIX + col + 16];
        } else {
            h16* o = (h16*)outp;
            float bb = (MODE == 1) ? aux[row] : 0.f;
            o[(size_t)row * NPIX + col]      = (h16)(c0[r] + bb);
            o[(size_t)row * NPIX + col + 16] = (h16)(c1[r] + bb);
        }
    }
}

// ---------------- per-head gram matrix G[h] = q[h] * k[h]^T (split-K atomics) ----------------
// 128 thr = 4 waves; wave (mt, nt) owns one 16x16 tile of the 32x32 result.
__global__ void gram_wmma(const h16* __restrict__ qkv, float* __restrict__ G) {
    const int h    = blockIdx.y;
    const int tid  = threadIdx.x;
    const int lane = tid & 31, l = lane & 15, hi = lane >> 4;
    const int w    = tid >> 5, mt = w & 1, nt = w >> 1;
    const size_t qrow = (size_t)(h * 32 + mt * 16 + l);
    const size_t krow = (size_t)(CCH + h * 32 + nt * 16 + l);
    const int pbase = blockIdx.x * 2048;

    v8f c = {};
    for (int pi = 0; pi < 2048; pi += 32) {
        const int p = pbase + pi;
        v16h a, b;
        const v8h alo = *(const v8h*)&qkv[qrow * NPIX + p + hi * 16];
        const v8h ahi = *(const v8h*)&qkv[qrow * NPIX + p + hi * 16 + 8];
        const v8h blo = *(const v8h*)&qkv[krow * NPIX + p + hi * 16];
        const v8h bhi = *(const v8h*)&qkv[krow * NPIX + p + hi * 16 + 8];
        #pragma unroll
        for (int i = 0; i < 8; ++i) { a[i] = alo[i]; a[i + 8] = ahi[i];
                                      b[i] = blo[i]; b[i + 8] = bhi[i]; }
        c = __builtin_amdgcn_wmma_f32_16x16x32_f16(false, a, false, b, (short)0, c, false, false);
    }
    #pragma unroll
    for (int r = 0; r < 8; ++r)
        atomicAdd(&G[h * 1024 + (mt * 16 + r + hi * 8) * 32 + nt * 16 + l], c[r]);
}

// ---------------- per-row sum of squares for q,k rows (384 rows) ----------------
__global__ void rownorm_kernel(const h16* __restrict__ qkv, float* __restrict__ norm2) {
    __shared__ float sd[256];
    const int row  = blockIdx.x;              // 0..383
    const int base = blockIdx.y * (NPIX / 8); // 8 chunks of 25088
    float acc = 0.f;
    for (int i = threadIdx.x; i < NPIX / 8; i += 256) {
        float v = (float)qkv[(size_t)row * NPIX + base + i];
        acc += v * v;
    }
    sd[threadIdx.x] = acc; __syncthreads();
    for (int s = 128; s > 0; s >>= 1) {
        if (threadIdx.x < s) sd[threadIdx.x] += sd[threadIdx.x + s];
        __syncthreads();
    }
    if (threadIdx.x == 0) atomicAdd(&norm2[row], sd[0]);
}

// ---------------- fold norms + mn_w + gating into per-head 32x32 gm ----------------
__global__ void make_gm(const float* __restrict__ G, const float* __restrict__ norm2,
                        const float* __restrict__ mnw, const float* __restrict__ gating,
                        float* __restrict__ gm) {
    __shared__ float s_attn[1024];
    __shared__ float s_nq[32], s_nk[32];
    const int h = blockIdx.x, t = threadIdx.x;   // 1024 threads
    if (t < 32)        s_nq[t]      = fmaxf(sqrtf(norm2[h * 32 + t]), 1e-12f);
    else if (t < 64)   s_nk[t - 32] = fmaxf(sqrtf(norm2[CCH + h * 32 + (t - 32)]), 1e-12f);
    __syncthreads();
    s_attn[t] = G[h * 1024 + t] / (s_nq[t >> 5] * s_nk[t & 31]);
    __syncthreads();
    const int d = t >> 5, e2 = t & 31;
    const float g1 = gating[h], g2 = gating[NHD + h];
    float acc = 0.f;
    for (int e = 0; e < 32; ++e)
        acc += s_attn[d * 32 + e] * (g1 * mnw[e2 * 32 + e] + g2 * mnw[(32 + e2) * 32 + e]);
    gm[h * 1024 + d * 32 + e2] = acc;
}

// ---------------- fused mi = gm@illu, *v, softmax over d, f16 out ----------------
__global__ void mi_softmax(const float* __restrict__ cond, const float* __restrict__ gm,
                           const h16* __restrict__ qkv, h16* __restrict__ out) {
    __shared__ float sgm[NHD * 1024];  // 24 KB
    for (int i = threadIdx.x; i < NHD * 1024; i += 256) sgm[i] = gm[i];
    __syncthreads();
    const int p = blockIdx.x * 256 + threadIdx.x;
    for (int h = 0; h < NHD; ++h) {
        float il[32];
        #pragma unroll
        for (int e = 0; e < 32; ++e) il[e] = cond[(size_t)(h * 32 + e) * NPIX + p];
        float s[32]; float mx = -3.4e38f;
        #pragma unroll
        for (int d = 0; d < 32; ++d) {
            float acc = 0.f;
            #pragma unroll
            for (int e = 0; e < 32; ++e) acc += sgm[h * 1024 + d * 32 + e] * il[e];
            float vv = (float)qkv[(size_t)(2 * CCH + h * 32 + d) * NPIX + p];
            float t = acc * vv;
            s[d] = t; mx = fmaxf(mx, t);
        }
        float sum = 0.f;
        #pragma unroll
        for (int d = 0; d < 32; ++d) { float e0 = __expf(s[d] - mx); s[d] = e0; sum += e0; }
        float inv = 1.f / sum;
        #pragma unroll
        for (int d = 0; d < 32; ++d)
            out[(size_t)(h * 32 + d) * NPIX + p] = (h16)(s[d] * inv);
    }
}

// ---------------- launcher ----------------
extern "C" void kernel_launch(void* const* d_in, const int* in_sizes, int n_in,
                              void* d_out, int out_size, void* d_ws, size_t ws_size,
                              hipStream_t stream) {
    (void)in_sizes; (void)n_in; (void)out_size; (void)ws_size;
    const float* x      = (const float*)d_in[0];
    const float* cond   = (const float*)d_in[1];
    const float* ln_g   = (const float*)d_in[2];
    const float* ln_b   = (const float*)d_in[3];
    const float* pre_w  = (const float*)d_in[4];
    const float* pre_b  = (const float*)d_in[5];
    const float* qkv_w  = (const float*)d_in[6];
    const float* mn_w   = (const float*)d_in[7];
    const float* gating = (const float*)d_in[8];
    const float* proj_w = (const float*)d_in[9];
    float* out = (float*)d_out;

    char* ws = (char*)d_ws;
    h16*   preW16  = (h16*)(ws + 0);          // 192*192 f16
    h16*   qkvW16  = (h16*)(ws + 73728);      // 576*192 f16
    h16*   projW16 = (h16*)(ws + 294912);     // 192*192 f16
    float* norm2   = (float*)(ws + 393216);   // 384 f32
    float* G       = (float*)(ws + 395264);   // 6*32*32 f32
    float* gm      = (float*)(ws + 419840);   // 6*32*32 f32
    const size_t PLANE = (size_t)CCH * NPIX * sizeof(h16);   // 77,070,336 B
    const size_t DB = 1ull << 20;
    h16* xn  = (h16*)(ws + DB);                 // [192,N] f16 (reused as softmax out)
    h16* y   = (h16*)(ws + DB + PLANE);         // [192,N] f16
    h16* qkv = (h16*)(ws + DB + 2 * PLANE);     // [576,N] f16

    // weight conversion + accumulator init
    cvt_f32_to_f16<<<(36864 + 255) / 256, 256, 0, stream>>>(pre_w,  preW16, 36864);
    cvt_f32_to_f16<<<(110592 + 255) / 256, 256, 0, stream>>>(qkv_w, qkvW16, 110592);
    cvt_f32_to_f16<<<(36864 + 255) / 256, 256, 0, stream>>>(proj_w, projW16, 36864);
    zero_f32<<<2, 256, 0, stream>>>(norm2, 384);
    zero_f32<<<24, 256, 0, stream>>>(G, 6144);

    // 1) LayerNorm -> xn (f16)
    layernorm_kernel<<<NPIX / 256, 256, 0, stream>>>(x, ln_g, ln_b, xn);

    // 2) pre: y = pre_w @ xn + pre_b
    {
        dim3 g(NPIX / 64, 192 / 64);
        gemm_wmma<1><<<g, 256, 0, stream>>>(preW16, xn, pre_b, (void*)y, CCH);
    }
    // 3) qkv = qkv_w @ y
    {
        dim3 g(NPIX / 64, 576 / 64);
        gemm_wmma<0><<<g, 256, 0, stream>>>(qkvW16, y, nullptr, (void*)qkv, CCH);
    }
    // 4) row sum-of-squares for q,k rows
    {
        dim3 g(2 * CCH, 8);
        rownorm_kernel<<<g, 256, 0, stream>>>(qkv, norm2);
    }
    // 5) per-head gram matrices (split-K WMMA + atomics)
    {
        dim3 g(NPIX / 2048, NHD);
        gram_wmma<<<g, 128, 0, stream>>>(qkv, G);
    }
    // 6) fold normalization, mn_w, gating into gm
    make_gm<<<NHD, 1024, 0, stream>>>(G, norm2, mn_w, gating, gm);

    // 7) mi -> *v -> softmax -> f16 (reuse xn buffer)
    mi_softmax<<<NPIX / 256, 256, 0, stream>>>(cond, gm, qkv, xn);

    // 8) proj + residual -> f32 d_out
    {
        dim3 g(NPIX / 64, 192 / 64);
        gemm_wmma<2><<<g, 256, 0, stream>>>(projW16, xn, x, (void*)out, CCH);
    }
}
